// MultiHeadAttention_60524679135258
// MI455X (gfx1250) — matrix operand order repaired and checked by reading, not compiled, not run on hardware
//
#include <hip/hip_runtime.h>
#include <hip/hip_bf16.h>

typedef _Float16 hf;
typedef __attribute__((ext_vector_type(16))) _Float16 v16h;
typedef __attribute__((ext_vector_type(8)))  _Float16 v8h;
typedef __attribute__((ext_vector_type(8)))  float    v8f;

union AF { v16h v; v8h h[2]; };

__device__ inline v8h  ld8 (const hf* p) { return *(const v8h*)p; }
__device__ inline v16h ld16(const hf* p) { return *(const v16h*)p; }

__device__ inline v8f wmma_f16(v16h a, v16h b, v8f c) {
  // D = A(16x32 f16) * B(32x16 f16) + C(16x16 f32)
  return __builtin_amdgcn_wmma_f32_16x16x32_f16(
      false, a, false, b, (short)0, c, false, false);
}

// ---------------------------------------------------------------------------
// fp32 -> fp16 conversion (memory bound)
// ---------------------------------------------------------------------------
__global__ void cvt_f32_f16(const float* __restrict__ src, hf* __restrict__ dst, int n) {
  int i = blockIdx.x * blockDim.x + threadIdx.x;
  if (i < n) dst[i] = (hf)src[i];
}

// ---------------------------------------------------------------------------
// QKV projection:  out = X(f16)[4096x1024] @ W(f16)^T[1024x1024] + bias
// z = 0: Q (scaled by 1/8, -> Qp [B,H,S,64])
// z = 1: K (-> Kp [B,H,S,64])
// z = 2: V (-> Vt [B,H,64,S]  i.e. transposed per head)
// Block: 256 threads = 8 waves; block tile 128x128; wave tile 32x64.
// Software-pipelined: chunk k+1 loads issue before chunk k WMMAs.
// ---------------------------------------------------------------------------
__global__ __launch_bounds__(256) void qkv_gemm(
    const hf* __restrict__ qf, const hf* __restrict__ kf, const hf* __restrict__ vf,
    const hf* __restrict__ wq, const hf* __restrict__ wk, const hf* __restrict__ wv,
    const float* __restrict__ bq, const float* __restrict__ bk, const float* __restrict__ bv,
    hf* __restrict__ Qp, hf* __restrict__ Kp, hf* __restrict__ Vt) {
  const int z = blockIdx.z;
  const hf* X = (z == 0) ? qf : (z == 1) ? kf : vf;
  const hf* W = (z == 0) ? wq : (z == 1) ? wk : wv;
  const float* bias = (z == 0) ? bq : (z == 1) ? bk : bv;

  const int lane = threadIdx.x & 31;
  const int wid  = threadIdx.x >> 5;       // 0..7
  const int wm   = wid >> 1;               // 0..3
  const int wn   = wid & 1;                // 0..1
  const int Mb   = blockIdx.y * 128 + wm * 32;
  const int Nb   = blockIdx.x * 128 + wn * 64;
  const int lc   = lane & 15;
  const int hi   = lane >> 4;              // half-wave id
  const int abase = hi ? 16 : 0;           // A frag K base (elements)
  const int bbase = hi ? 16 : 0;           // B frag K base (elements)

  v8f c[2][4] = {};

  const hf* arow0 = X + (size_t)(Mb + lc) * 1024;
  const hf* arow1 = X + (size_t)(Mb + 16 + lc) * 1024;
  const hf* brow[4];
#pragma unroll
  for (int j = 0; j < 4; ++j)
    brow[j] = W + (size_t)(Nb + j * 16 + lc) * 1024;

  AF a0[2], a1[2];
  v16h b[2][4];
  // preload chunk 0
  a0[0].h[0] = ld8(arow0 + abase);
  a0[0].h[1] = ld8(arow0 + abase + 8);
  a1[0].h[0] = ld8(arow1 + abase);
  a1[0].h[1] = ld8(arow1 + abase + 8);
#pragma unroll
  for (int j = 0; j < 4; ++j) b[0][j] = ld16(brow[j] + bbase);

  for (int kc = 0; kc < 1024; kc += 32) {
    const int cur = (kc >> 5) & 1;
    const int nxt = cur ^ 1;
    if (kc + 32 < 1024) {
      const int kn = kc + 32;
      a0[nxt].h[0] = ld8(arow0 + kn + abase);
      a0[nxt].h[1] = ld8(arow0 + kn + abase + 8);
      a1[nxt].h[0] = ld8(arow1 + kn + abase);
      a1[nxt].h[1] = ld8(arow1 + kn + abase + 8);
#pragma unroll
      for (int j = 0; j < 4; ++j) b[nxt][j] = ld16(brow[j] + kn + bbase);
      __builtin_prefetch(arow0 + kn + 256, 0, 3);   // WGP-scope prefetch
      __builtin_prefetch(brow[0] + kn + 256, 0, 3);
    }
#pragma unroll
    for (int j = 0; j < 4; ++j) {
      c[0][j] = wmma_f16(a0[cur].v, b[cur][j], c[0][j]);
      c[1][j] = wmma_f16(a1[cur].v, b[cur][j], c[1][j]);
    }
  }

#pragma unroll
  for (int i = 0; i < 2; ++i)
#pragma unroll
    for (int j = 0; j < 4; ++j)
#pragma unroll
      for (int r = 0; r < 8; ++r) {
        int m = Mb + i * 16 + r + hi * 8;       // global row (b*S + s)
        int n = Nb + j * 16 + lc;               // global col (h*64 + d)
        float val = c[i][j][r] + bias[n];
        int bb = m >> 11, s = m & 2047;
        int hh = n >> 6,  d = n & 63;
        size_t head = (size_t)(bb * 16 + hh);
        if (z == 0)      Qp[(head * 2048 + s) * 64 + d] = (hf)(val * 0.125f);
        else if (z == 1) Kp[(head * 2048 + s) * 64 + d] = (hf)val;
        else             Vt[(head * 64 + d) * 2048 + s] = (hf)val;
      }
}

// ---------------------------------------------------------------------------
// Causal flash attention.
// Grid: (qblock=16, h=16, b=2). Block: 256 threads = 8 waves.
// Wave w handles q rows [q0 + 16w, q0 + 16w + 15]; keys streamed in 32-chunks.
// K^T fragments double-buffered; V fragments issued early so their latency
// hides behind the softmax VALU work and the LDS P-transpose round-trip.
// ---------------------------------------------------------------------------
__global__ __launch_bounds__(256) void attn_kernel(
    const hf* __restrict__ Qp, const hf* __restrict__ Kp,
    const hf* __restrict__ Vt, hf* __restrict__ AO) {
  __shared__ __align__(64) hf lds[8][16][32];   // per-wave P transpose buffer

  const int lane = threadIdx.x & 31;
  const int w    = threadIdx.x >> 5;
  const int lc   = lane & 15;
  const int hi   = lane >> 4;
  const int abase = hi ? 16 : 0;
  const int bbase = hi ? 16 : 0;

  const int q0 = blockIdx.x * 128;
  const int h  = blockIdx.y;
  const int b  = blockIdx.z;
  const size_t head = (size_t)(b * 16 + h);
  const hf* Qb = Qp + head * (2048 * 64);
  const hf* Kb = Kp + head * (2048 * 64);
  const hf* Vb = Vt + head * (64 * 2048);

  const int qrow0 = q0 + w * 16;

  // Q fragments resident in registers: A0 = dims 0..31, A1 = dims 32..63
  const hf* qr = Qb + (size_t)(qrow0 + lc) * 64;
  AF A0, A1;
  A0.h[0] = ld8(qr + abase);       A0.h[1] = ld8(qr + abase + 8);
  A1.h[0] = ld8(qr + 32 + abase);  A1.h[1] = ld8(qr + 32 + abase + 8);

  v8f acc[4] = {};
  float mrun[8], lrun[8];
#pragma unroll
  for (int r = 0; r < 8; ++r) { mrun[r] = -1e30f; lrun[r] = 0.0f; }

  const int lastk = qrow0 + 15;                 // causal: highest visible key
  const int nch   = (lastk + 32) >> 5;          // ceil((lastk+1)/32)

  // K^T fragment buffers: [buf][tile(0..1)][kdim half(0..1)]
  v16h kfr[2][2][2];
  {
    const hf* kr0 = Kb + (size_t)(lc)*64;
    const hf* kr1 = Kb + (size_t)(16 + lc) * 64;
    kfr[0][0][0] = ld16(kr0 + bbase);
    kfr[0][0][1] = ld16(kr0 + 32 + bbase);
    kfr[0][1][0] = ld16(kr1 + bbase);
    kfr[0][1][1] = ld16(kr1 + 32 + bbase);
  }

  for (int cc = 0; cc < nch; ++cc) {
    const int kb  = cc * 32;
    const int cur = cc & 1;
    const int nxt = cur ^ 1;
    if (cc + 1 < nch) {
      const int kn = kb + 32;
      const hf* kr0 = Kb + (size_t)(kn + lc) * 64;
      const hf* kr1 = Kb + (size_t)(kn + 16 + lc) * 64;
      kfr[nxt][0][0] = ld16(kr0 + bbase);
      kfr[nxt][0][1] = ld16(kr0 + 32 + bbase);
      kfr[nxt][1][0] = ld16(kr1 + bbase);
      kfr[nxt][1][1] = ld16(kr1 + 32 + bbase);
    }

    v8f s0 = {}, s1 = {};
    s0 = wmma_f16(A0.v, kfr[cur][0][0], s0);
    s0 = wmma_f16(A1.v, kfr[cur][0][1], s0);
    s1 = wmma_f16(A0.v, kfr[cur][1][0], s1);
    s1 = wmma_f16(A1.v, kfr[cur][1][1], s1);

    // issue V fragments now; consumed by the P@V WMMAs at the end of chunk
    v16h vfrag[4];
#pragma unroll
    for (int j = 0; j < 4; ++j)
      vfrag[j] = ld16(Vb + (size_t)(j * 16 + lc) * 2048 + kb + bbase);

    float mnew[8], alpha[8], p0[8], p1[8];
#pragma unroll
    for (int r = 0; r < 8; ++r) {
      int row = qrow0 + r + hi * 8;
      if (kb + lc > row)      s0[r] = -1e9f;
      if (kb + 16 + lc > row) s1[r] = -1e9f;
    }
#pragma unroll
    for (int r = 0; r < 8; ++r) {
      float mx = fmaxf(s0[r], s1[r]);
#pragma unroll
      for (int msk = 1; msk < 16; msk <<= 1) mx = fmaxf(mx, __shfl_xor(mx, msk, 32));
      mnew[r]  = fmaxf(mrun[r], mx);
      alpha[r] = __expf(mrun[r] - mnew[r]);
      mrun[r]  = mnew[r];
    }
#pragma unroll
    for (int r = 0; r < 8; ++r) {
      p0[r] = __expf(s0[r] - mnew[r]);
      p1[r] = __expf(s1[r] - mnew[r]);
      float sm = p0[r] + p1[r];
#pragma unroll
      for (int msk = 1; msk < 16; msk <<= 1) sm += __shfl_xor(sm, msk, 32);
      lrun[r] = lrun[r] * alpha[r] + sm;
    }

    // C-layout -> A-layout for P via wave-private LDS round-trip
#pragma unroll
    for (int r = 0; r < 8; ++r) {
      lds[w][r + hi * 8][lc]      = (hf)p0[r];
      lds[w][r + hi * 8][16 + lc] = (hf)p1[r];
    }
    asm volatile("s_wait_dscnt 0" ::: "memory");
    AF P;
    P.h[0] = *(const v8h*)&lds[w][lc][abase];
    P.h[1] = *(const v8h*)&lds[w][lc][abase + 8];

    // rescale running accumulator, then acc += P @ V_chunk
#pragma unroll
    for (int j = 0; j < 4; ++j)
#pragma unroll
      for (int r = 0; r < 8; ++r) acc[j][r] *= alpha[r];
#pragma unroll
    for (int j = 0; j < 4; ++j)
      acc[j] = wmma_f16(P.v, vfrag[j], acc[j]);
  }

  // epilogue: normalize and write head-merged [B, S, 1024] f16 buffer
#pragma unroll
  for (int j = 0; j < 4; ++j)
#pragma unroll
    for (int r = 0; r < 8; ++r) {
      int s = qrow0 + r + hi * 8;
      int d = j * 16 + lc;
      AO[((size_t)b * 2048 + s) * 1024 + h * 64 + d] = (hf)(acc[j][r] / lrun[r]);
    }
}

// ---------------------------------------------------------------------------
// Output projection: d_out = AO(f16)[4096x1024] @ Wo(f16)^T + b_o   (fp32 out)
// ---------------------------------------------------------------------------
__global__ __launch_bounds__(256) void out_gemm(
    const hf* __restrict__ AO, const hf* __restrict__ Wo,
    const float* __restrict__ bo, float* __restrict__ out) {
  const int lane = threadIdx.x & 31;
  const int wid  = threadIdx.x >> 5;
  const int wm = wid >> 1, wn = wid & 1;
  const int Mb = blockIdx.y * 128 + wm * 32;
  const int Nb = blockIdx.x * 128 + wn * 64;
  const int lc = lane & 15;
  const int hi = lane >> 4;
  const int abase = hi ? 16 : 0;
  const int bbase = hi ? 16 : 0;

  v8f c[2][4] = {};
  const hf* arow0 = AO + (size_t)(Mb + lc) * 1024;
  const hf* arow1 = AO + (size_t)(Mb + 16 + lc) * 1024;
  const hf* brow[4];
#pragma unroll
  for (int j = 0; j < 4; ++j)
    brow[j] = Wo + (size_t)(Nb + j * 16 + lc) * 1024;

  AF a0[2], a1[2];
  v16h b[2][4];
  a0[0].h[0] = ld8(arow0 + abase);
  a0[0].h[1] = ld8(arow0 + abase + 8);
  a1[0].h[0] = ld8(arow1 + abase);
  a1[0].h[1] = ld8(arow1 + abase + 8);
#pragma unroll
  for (int j = 0; j < 4; ++j) b[0][j] = ld16(brow[j] + bbase);

  for (int kc = 0; kc < 1024; kc += 32) {
    const int cur = (kc >> 5) & 1;
    const int nxt = cur ^ 1;
    if (kc + 32 < 1024) {
      const int kn = kc + 32;
      a0[nxt].h[0] = ld8(arow0 + kn + abase);
      a0[nxt].h[1] = ld8(arow0 + kn + abase + 8);
      a1[nxt].h[0] = ld8(arow1 + kn + abase);
      a1[nxt].h[1] = ld8(arow1 + kn + abase + 8);
#pragma unroll
      for (int j = 0; j < 4; ++j) b[nxt][j] = ld16(brow[j] + kn + bbase);
      __builtin_prefetch(arow0 + kn + 256, 0, 3);
      __builtin_prefetch(brow[0] + kn + 256, 0, 3);
    }
#pragma unroll
    for (int j = 0; j < 4; ++j) {
      c[0][j] = wmma_f16(a0[cur].v, b[cur][j], c[0][j]);
      c[1][j] = wmma_f16(a1[cur].v, b[cur][j], c[1][j]);
    }
  }

#pragma unroll
  for (int i = 0; i < 2; ++i)
#pragma unroll
    for (int j = 0; j < 4; ++j)
#pragma unroll
      for (int r = 0; r < 8; ++r) {
        int m = Mb + i * 16 + r + hi * 8;
        int n = Nb + j * 16 + lc;
        out[(size_t)m * 1024 + n] = c[i][j][r] + bo[n];
      }
}

// ---------------------------------------------------------------------------
// Launcher
// ---------------------------------------------------------------------------
extern "C" void kernel_launch(void* const* d_in, const int* in_sizes, int n_in,
                              void* d_out, int out_size, void* d_ws, size_t ws_size,
                              hipStream_t stream) {
  const float* q   = (const float*)d_in[0];
  const float* k   = (const float*)d_in[1];
  const float* v   = (const float*)d_in[2];
  // d_in[3] = causal mask: implemented analytically, unused
  const float* w_q = (const float*)d_in[4];
  const float* b_q = (const float*)d_in[5];
  const float* w_k = (const float*)d_in[6];
  const float* b_k = (const float*)d_in[7];
  const float* w_v = (const float*)d_in[8];
  const float* b_v = (const float*)d_in[9];
  const float* w_o = (const float*)d_in[10];
  const float* b_o = (const float*)d_in[11];

  const size_t M4 = (size_t)4096 * 1024;   // 4M elements
  const size_t M1 = (size_t)1024 * 1024;   // 1M elements
  hf* ws  = (hf*)d_ws;
  hf* qf  = ws;                // [4096,1024] f16
  hf* kf  = ws + 1 * M4;
  hf* vf  = ws + 2 * M4;
  hf* wqh = ws + 3 * M4;
  hf* wkh = wqh + M1;
  hf* wvh = wkh + M1;
  hf* woh = wvh + M1;
  hf* Qp  = woh + M1;          // [B,H,S,64]
  hf* Kp  = Qp + M4;           // [B,H,S,64]
  hf* Vt  = Kp + M4;           // [B,H,64,S]
  hf* AO  = Vt + M4;           // [B,S,1024]

  const int thr = 256;
  cvt_f32_f16<<<(int)((M4 + thr - 1) / thr), thr, 0, stream>>>(q, qf, (int)M4);
  cvt_f32_f16<<<(int)((M4 + thr - 1) / thr), thr, 0, stream>>>(k, kf, (int)M4);
  cvt_f32_f16<<<(int)((M4 + thr - 1) / thr), thr, 0, stream>>>(v, vf, (int)M4);
  cvt_f32_f16<<<(int)((M1 + thr - 1) / thr), thr, 0, stream>>>(w_q, wqh, (int)M1);
  cvt_f32_f16<<<(int)((M1 + thr - 1) / thr), thr, 0, stream>>>(w_k, wkh, (int)M1);
  cvt_f32_f16<<<(int)((M1 + thr - 1) / thr), thr, 0, stream>>>(w_v, wvh, (int)M1);
  cvt_f32_f16<<<(int)((M1 + thr - 1) / thr), thr, 0, stream>>>(w_o, woh, (int)M1);

  qkv_gemm<<<dim3(8, 32, 3), 256, 0, stream>>>(qf, kf, vf, wqh, wkh, wvh,
                                               b_q, b_k, b_v, Qp, Kp, Vt);

  attn_kernel<<<dim3(16, 16, 2), 256, 0, stream>>>(Qp, Kp, Vt, AO);

  out_gemm<<<dim3(8, 32, 1), 256, 0, stream>>>(AO, woh, b_o, (float*)d_out);
}
